// ArcFaceLoss_84284438216806
// MI455X (gfx1250) — matrix operand order repaired and checked by reading, not compiled, not run on hardware
//
#include <hip/hip_runtime.h>
#include <math.h>
#include <stdint.h>

// ---- problem constants -------------------------------------------------
#define B_ROWS 2048
#define D_DIM  512
#define V_CLS  100000
#define NCH    1563                 // ceil(V / 64)
#define VPAD   (NCH * 64)           // 100032
#define S_SCALE 64.0f
#define COS_M_C 0.87758256189037271f
#define SIN_M_C 0.47942553860420301f
#define MM_C    0.23971276930210150f
#define THRESH_C (-0.87758256189037271f)

typedef __attribute__((ext_vector_type(16))) _Float16 v16h;
typedef __attribute__((ext_vector_type(8)))  _Float16 v8h;
typedef __attribute__((ext_vector_type(4)))  _Float16 v4h;
typedef __attribute__((ext_vector_type(8)))  float    v8f;

union FragH { v16h v; v8h h[2]; };

// ---- CDNA5 async global->LDS copy, raw ISA (portable across toolchains) --
// Emits exactly what the compiler emits for the builtin:
//   global_load_async_to_lds_b128 vDST(lds addr), v[ADDR:ADDR+1], off
// Volatile asm keeps strict program order vs. the barrier blocks below.
__device__ __forceinline__ void async_cp16(const _Float16* g, const _Float16* l)
{
    const uint32_t loff = (uint32_t)(uintptr_t)l;   // low 32 bits = LDS offset
    asm volatile("global_load_async_to_lds_b128 %0, %1, off"
                 :: "v"(loff), "v"((uint64_t)(uintptr_t)g)
                 : "memory");
}

// Wait for this wave's async copies + workgroup barrier, fused into one
// un-breakable asm block.
//  - input  `dep` (prev iteration's last WMMA result) pins all of tile k's
//    loads/WMMAs ABOVE barrier k+1 (they can't sink into tile k+2's copies)
//  - output `tok` (=0), added to every LDS fragment address, pins tile k's
//    loads BELOW barrier k (they can't rise above the asynccnt wait)
__device__ __forceinline__ uint32_t sync_tile_barrier(float dep)
{
    uint32_t tok;
    asm volatile(
        "s_wait_dscnt 0x0\n\t"
        "s_wait_asynccnt 0x0\n\t"
        "s_barrier_signal -1\n\t"
        "s_barrier_wait -1\n\t"
        "s_mov_b32 %0, 0"
        : "=s"(tok) : "v"(dep) : "memory");
    return tok;
}

// ---- kernel 1: row L2-normalize f32 -> f16 -----------------------------
__global__ __launch_bounds__(128)
void prep_rows(const float* __restrict__ src, _Float16* __restrict__ dst,
               int validRows)
{
    const int row = blockIdx.x;
    const int tid = threadIdx.x;
    if (row >= validRows) {
        v4h z = (v4h){(_Float16)0, (_Float16)0, (_Float16)0, (_Float16)0};
        ((v4h*)(dst + (size_t)row * D_DIM))[tid] = z;
        return;
    }
    const float4 x = ((const float4*)(src + (size_t)row * D_DIM))[tid];
    float ss = x.x * x.x + x.y * x.y + x.z * x.z + x.w * x.w;
#pragma unroll
    for (int d = 16; d >= 1; d >>= 1) ss += __shfl_xor(ss, d, 32);
    __shared__ float wsum[4];
    const int wv = tid >> 5, lane = tid & 31;
    if (lane == 0) wsum[wv] = ss;
    __syncthreads();
    const float tot = wsum[0] + wsum[1] + wsum[2] + wsum[3];
    const float inv = 1.0f / fmaxf(sqrtf(tot), 1e-12f);
    v4h o;
    o.x = (_Float16)(x.x * inv);
    o.y = (_Float16)(x.y * inv);
    o.z = (_Float16)(x.z * inv);
    o.w = (_Float16)(x.w * inv);
    ((v4h*)(dst + (size_t)row * D_DIM))[tid] = o;
}

// ---- kernel 2: fused WMMA GEMM + margin + chunk softmax partials -------
// grid (NCH, B/128); block 256 = 8 waves. Tile: 128 B-rows x 64 V-cols.
// wave w owns rows [128*by + 16*w, +16), all 64 cols -> softmax stats are
// wave-local. Double-buffered LDS, async copy overlapped with WMMA.
#define LDS_STRIDE 40   // halves per row (padded vs 32 to spread banks)

__global__ __launch_bounds__(256)
void arc_gemm(const _Float16* __restrict__ eF, const _Float16* __restrict__ wF,
              const int* __restrict__ labels,
              float* __restrict__ mPart, float* __restrict__ sPart,
              float* __restrict__ tgt)
{
    __shared__ __align__(16) _Float16 aT[2][128 * LDS_STRIDE];
    __shared__ __align__(16) _Float16 bT[2][64 * LDS_STRIDE];

    const int tid   = threadIdx.x;
    const int chunk = blockIdx.x;
    const int mBase = blockIdx.y * 128;
    const int nBase = chunk * 64;
    const int wv    = tid >> 5;
    const int lane  = tid & 31;
    const int lg    = lane & 15;
    const int lh    = lane >> 4;

    // per-thread staging coordinates (A: 2 chunks, B: 1 chunk of 16B)
    const int aR0 = (tid) >> 2,        aC0 = (tid) & 3;
    const int aR1 = (tid + 256) >> 2,  aC1 = (tid + 256) & 3;
    const int bR  = tid >> 2,          bC  = tid & 3;

    const _Float16* gA0 = eF + (size_t)(mBase + aR0) * D_DIM + aC0 * 8;
    const _Float16* gA1 = eF + (size_t)(mBase + aR1) * D_DIM + aC1 * 8;
    const _Float16* gB  = wF + (size_t)(nBase + bR)  * D_DIM + bC  * 8;
    const int lA0 = aR0 * LDS_STRIDE + aC0 * 8;
    const int lA1 = aR1 * LDS_STRIDE + aC1 * 8;
    const int lB  = bR  * LDS_STRIDE + bC  * 8;

    v8f acc[4];
#pragma unroll
    for (int i = 0; i < 4; ++i) acc[i] = (v8f){0.f,0.f,0.f,0.f,0.f,0.f,0.f,0.f};

    // ---------------- async-copy double-buffered pipeline ----------------
    // prologue: start tile 0 into buffer 0 (volatile -> stays before barrier 0)
    async_cp16(gA0, aT[0] + lA0);
    async_cp16(gA1, aT[0] + lA1);
    async_cp16(gB,  bT[0] + lB);

    float dep = 0.0f;
    for (int ks = 0; ks < 16; ++ks) {
        const int cur = ks & 1;
        // tile ks landed; token pins fragment loads below this barrier,
        // dep pins previous tile's loads/WMMAs above it.
        const uint32_t tok = sync_tile_barrier(dep);
        if (ks < 15) {
            const int kk = (ks + 1) * 32;
            async_cp16(gA0 + kk, aT[cur ^ 1] + lA0);
            async_cp16(gA1 + kk, aT[cur ^ 1] + lA1);
            async_cp16(gB  + kk, bT[cur ^ 1] + lB);
        }
        // compute from buf[cur]
        FragH a;
        const v8h* ap = (const v8h*)(aT[cur] + tok + (wv * 16 + lg) * LDS_STRIDE + lh * 8);
        a.h[0] = ap[0];
        a.h[1] = ap[2];
#pragma unroll
        for (int nt = 0; nt < 4; ++nt) {
            FragH b;
            const v8h* bp = (const v8h*)(bT[cur] + tok + (nt * 16 + lg) * LDS_STRIDE + lh * 8);
            b.h[0] = bp[0];
            b.h[1] = bp[2];
            acc[nt] = __builtin_amdgcn_wmma_f32_16x16x32_f16(
                false, a.v, false, b.v, (short)0, acc[nt], false, false);
        }
        dep = acc[3][0];   // scheduling edge into the next barrier
    }

    // epilogue: margin transform + per-row (max, sum-exp) over this chunk
    const int bRow0 = mBase + wv * 16;
#pragma unroll
    for (int g = 0; g < 8; ++g) {
        float l[4];
#pragma unroll
        for (int nt = 0; nt < 4; ++nt) {
            const float x  = acc[nt][g];
            const float ct = fminf(fmaxf(x, -1.f), 1.f);
            const float st = sqrtf(fmaxf(1.f - ct * ct, 0.f));
            float ctm = ct * COS_M_C - st * SIN_M_C;
            ctm = (ct > THRESH_C) ? ctm : (ct - MM_C);
            const int vcol = nBase + nt * 16 + lg;
            l[nt] = (vcol < V_CLS) ? (S_SCALE * ctm) : -INFINITY;
        }
        const int bg = bRow0 + g + 8 * lh;     // global B row for this (g, half)
        const int co = labels[bg] - nBase;     // target column offset, if in chunk
#pragma unroll
        for (int nt = 0; nt < 4; ++nt)
            if (co == nt * 16 + lg) tgt[bg] = l[nt];

        float mx = fmaxf(fmaxf(l[0], l[1]), fmaxf(l[2], l[3]));
#pragma unroll
        for (int d = 8; d >= 1; d >>= 1) mx = fmaxf(mx, __shfl_xor(mx, d, 32));
        float se = 0.f;
#pragma unroll
        for (int nt = 0; nt < 4; ++nt) se += __expf(l[nt] - mx);
#pragma unroll
        for (int d = 8; d >= 1; d >>= 1) se += __shfl_xor(se, d, 32);

        if (lg == 0) {
            const size_t idx = (size_t)bg * NCH + chunk;
            mPart[idx] = mx;
            sPart[idx] = se;
        }
    }
}

// ---- kernel 3: merge chunk partials -> per-row NLL ---------------------
__global__ __launch_bounds__(256)
void row_reduce(const float* __restrict__ mPart, const float* __restrict__ sPart,
                const float* __restrict__ tgt, float* __restrict__ nll)
{
    const int b = blockIdx.x;
    const int tid = threadIdx.x;
    __shared__ float red[256];
    const float* mrow = mPart + (size_t)b * NCH;
    const float* srow = sPart + (size_t)b * NCH;

    float mx = -INFINITY;
    for (int c = tid; c < NCH; c += 256) mx = fmaxf(mx, mrow[c]);
    red[tid] = mx; __syncthreads();
    for (int s = 128; s > 0; s >>= 1) {
        if (tid < s) red[tid] = fmaxf(red[tid], red[tid + s]);
        __syncthreads();
    }
    const float M = red[0];
    __syncthreads();

    float ss = 0.f;
    for (int c = tid; c < NCH; c += 256) ss += srow[c] * __expf(mrow[c] - M);
    red[tid] = ss; __syncthreads();
    for (int s = 128; s > 0; s >>= 1) {
        if (tid < s) red[tid] += red[tid + s];
        __syncthreads();
    }
    if (tid == 0) nll[b] = M + logf(red[0]) - tgt[b];
}

// ---- kernel 4: deterministic mean over B -------------------------------
__global__ __launch_bounds__(256)
void mean_kernel(const float* __restrict__ nll, float* __restrict__ out)
{
    __shared__ float red[256];
    const int tid = threadIdx.x;
    float s = 0.f;
    for (int b = tid; b < B_ROWS; b += 256) s += nll[b];
    red[tid] = s; __syncthreads();
    for (int st = 128; st > 0; st >>= 1) {
        if (tid < st) red[tid] += red[tid + st];
        __syncthreads();
    }
    if (tid == 0) out[0] = red[0] * (1.0f / (float)B_ROWS);
}

// ---- host launcher -----------------------------------------------------
static inline size_t align256(size_t x) { return (x + 255) & ~(size_t)255; }

extern "C" void kernel_launch(void* const* d_in, const int* in_sizes, int n_in,
                              void* d_out, int out_size, void* d_ws, size_t ws_size,
                              hipStream_t stream)
{
    const float* emb    = (const float*)d_in[0];   // [B, D] f32
    const int*   labels = (const int*)  d_in[1];   // [B] int32
    const float* weight = (const float*)d_in[2];   // [V, D] f32

    char* ws = (char*)d_ws;
    size_t off = 0;
    _Float16* eF = (_Float16*)(ws + off); off = align256(off + (size_t)B_ROWS * D_DIM * 2);
    _Float16* wF = (_Float16*)(ws + off); off = align256(off + (size_t)VPAD * D_DIM * 2);
    float* mPart = (float*)(ws + off);    off = align256(off + (size_t)B_ROWS * NCH * 4);
    float* sPart = (float*)(ws + off);    off = align256(off + (size_t)B_ROWS * NCH * 4);
    float* tgtL  = (float*)(ws + off);    off = align256(off + (size_t)B_ROWS * 4);
    float* nll   = (float*)(ws + off);    off = align256(off + (size_t)B_ROWS * 4);

    prep_rows<<<B_ROWS, 128, 0, stream>>>(emb, eF, B_ROWS);
    prep_rows<<<VPAD,   128, 0, stream>>>(weight, wF, V_CLS);

    dim3 grid(NCH, B_ROWS / 128);
    arc_gemm<<<grid, 256, 0, stream>>>(eF, wF, labels, mPart, sPart, tgtL);

    row_reduce<<<B_ROWS, 256, 0, stream>>>(mPart, sPart, tgtL, nll);
    mean_kernel<<<1, 256, 0, stream>>>(nll, (float*)d_out);
}
